// model_17927193494376
// MI455X (gfx1250) — compile-verified, matrix operand order checked by reading, not run on hardware
//
#include <hip/hip_runtime.h>

// ---------------- model dimensions (fixed by the reference) ----------------
#define BB      16
#define SS      128
#define TT      128
#define HDIM    1024
#define VV      32000
#define LLAY    2
#define HIDE    512      // encoder per-direction hidden
#define SOS_TOK 3

// ---------------- CDNA5 WMMA types ----------------
typedef __attribute__((ext_vector_type(16))) __bf16 v16bf;
typedef __attribute__((ext_vector_type(8)))  __bf16 bf16x8;
typedef __attribute__((ext_vector_type(8)))  float  v8f;

__device__ __forceinline__ v8f wmma_bf16_acc(v16bf a, v16bf b, v8f c) {
  // D = A(16x32 bf16) * B(32x16 bf16) + C(16x16 f32)
  return __builtin_amdgcn_wmma_f32_16x16x32_bf16(false, a, false, b, (short)0, c,
                                                 false, false);
}

// Fragment loader (works for both A and B operands; identical K pattern):
//   lane l<16  -> row/col = l,     K = {k0..k0+7, k0+16..k0+23}
//   lane l>=16 -> row/col = l-16,  K = {k0+8..k0+15, k0+24..k0+31}
// kbase = k0 + (lane>=16)*8. Two contiguous 16-byte (8 x bf16) loads per lane.
__device__ __forceinline__ v16bf load_frag_bf16(const __bf16* __restrict__ row, int kbase) {
  bf16x8 lo = *(const bf16x8*)(row + kbase);
  bf16x8 hi = *(const bf16x8*)(row + kbase + 16);
  v16bf r;
#pragma unroll
  for (int e = 0; e < 8; ++e) { r[e] = lo[e]; r[8 + e] = hi[e]; }
  return r;
}

// Software-pipelined K-panel: one shared A stream vs 4 weight-row streams.
// Next iteration's fragments are loaded BEFORE the current iteration's WMMAs so
// the vmem latency overlaps XDL execution (critical at the low wave counts the
// recurrence imposes). K must be a multiple of 32 and >= 32.
__device__ __forceinline__ void gemm_panel4(const __bf16* __restrict__ arow,
                                            const __bf16* __restrict__ w0,
                                            const __bf16* __restrict__ w1,
                                            const __bf16* __restrict__ w2,
                                            const __bf16* __restrict__ w3,
                                            int K, int kb,
                                            v8f& a0, v8f& a1, v8f& a2, v8f& a3) {
  v16bf af = load_frag_bf16(arow, kb);
  v16bf b0 = load_frag_bf16(w0, kb);
  v16bf b1 = load_frag_bf16(w1, kb);
  v16bf b2 = load_frag_bf16(w2, kb);
  v16bf b3 = load_frag_bf16(w3, kb);
  for (int k0 = 32; k0 < K; k0 += 32) {
    v16bf afn = load_frag_bf16(arow, k0 + kb);
    v16bf n0  = load_frag_bf16(w0, k0 + kb);
    v16bf n1  = load_frag_bf16(w1, k0 + kb);
    v16bf n2  = load_frag_bf16(w2, k0 + kb);
    v16bf n3  = load_frag_bf16(w3, k0 + kb);
    a0 = wmma_bf16_acc(af, b0, a0);
    a1 = wmma_bf16_acc(af, b1, a1);
    a2 = wmma_bf16_acc(af, b2, a2);
    a3 = wmma_bf16_acc(af, b3, a3);
    af = afn; b0 = n0; b1 = n1; b2 = n2; b3 = n3;
  }
  a0 = wmma_bf16_acc(af, b0, a0);
  a1 = wmma_bf16_acc(af, b1, a1);
  a2 = wmma_bf16_acc(af, b2, a2);
  a3 = wmma_bf16_acc(af, b3, a3);
}

__device__ __forceinline__ float sigmoidf_(float x) { return 1.0f / (1.0f + __expf(-x)); }

// ---------------- fused LSTM step (gates GEMM via WMMA + cell update) ----------------
// gates[b][g*hid+j] = sum_k x[b][k] Wih[g*hid+j][k] + sum_k h[b][k] Whh[g*hid+j][k] + bias
// One wave owns 16 hidden units and the four gate tiles for them (A frag reused 4x).
__global__ void lstm_step_kernel(const __bf16* __restrict__ x, long x_row_stride, int Kx,
                                 const __bf16* __restrict__ hprev, int hid,
                                 const __bf16* __restrict__ Wih,
                                 const __bf16* __restrict__ Whh,
                                 const float* __restrict__ bias,
                                 float*  __restrict__ c_state,     // in-place (per-column safe)
                                 __bf16* __restrict__ h_out,       // ping-pong, next A operand
                                 __bf16* __restrict__ hs_bf,       // optional bf16 seq output
                                 float*  __restrict__ hs_f32,      // optional f32 seq output
                                 long hs_row_stride) {
  int wave = (int)((blockIdx.x * blockDim.x + threadIdx.x) >> 5);
  int lane = (int)(threadIdx.x & 31);
  if (wave >= (hid >> 4)) return;
  int jb   = wave << 4;
  int half = lane >> 4;
  int l16  = lane & 15;
  int kb   = half * 8;

  v8f acc0, acc1, acc2, acc3;
#pragma unroll
  for (int r = 0; r < 8; ++r) { acc0[r] = 0.0f; acc1[r] = 0.0f; acc2[r] = 0.0f; acc3[r] = 0.0f; }

  {
    const __bf16* xrow = x + (long)l16 * x_row_stride;
    long rw = (long)(jb + l16);
    gemm_panel4(xrow,
                Wih + (0 * (long)hid + rw) * Kx, Wih + (1 * (long)hid + rw) * Kx,
                Wih + (2 * (long)hid + rw) * Kx, Wih + (3 * (long)hid + rw) * Kx,
                Kx, kb, acc0, acc1, acc2, acc3);
  }
  {
    const __bf16* hrow = hprev + (long)l16 * hid;
    long rw = (long)(jb + l16);
    gemm_panel4(hrow,
                Whh + (0 * (long)hid + rw) * hid, Whh + (1 * (long)hid + rw) * hid,
                Whh + (2 * (long)hid + rw) * hid, Whh + (3 * (long)hid + rw) * hid,
                hid, kb, acc0, acc1, acc2, acc3);
  }

  int j = jb + l16;
  float bi = bias[0 * hid + j], bf = bias[1 * hid + j];
  float bg = bias[2 * hid + j], bo = bias[3 * hid + j];
#pragma unroll
  for (int r = 0; r < 8; ++r) {
    int m = r + half * 8;                      // C/D layout: VGPR r + half -> M
    float si = sigmoidf_(acc0[r] + bi);
    float sf = sigmoidf_(acc1[r] + bf);
    float tg = tanhf(acc2[r] + bg);
    float so = sigmoidf_(acc3[r] + bo);
    long  ci = (long)m * hid + j;
    float c2 = sf * c_state[ci] + si * tg;
    c_state[ci] = c2;
    float h2 = so * tanhf(c2);
    h_out[(long)m * hid + j] = (__bf16)h2;
    if (hs_bf)  hs_bf[(long)m * hs_row_stride + j]  = (__bf16)h2;
    if (hs_f32) hs_f32[(long)m * hs_row_stride + j] = h2;
  }
}

// ------- generic M=16 GEMM, 4 N-tiles per wave: C[16][N] = A[16][K] * W[N][K]^T -------
__global__ void gemm16x4_kernel(const __bf16* __restrict__ A, long a_row_stride, int K,
                                const __bf16* __restrict__ W,
                                const float* __restrict__ bias,
                                float*  __restrict__ Cf,
                                __bf16* __restrict__ Cbf,
                                long c_row_stride, long c_off,
                                int N, int act) {
  int wave = (int)((blockIdx.x * blockDim.x + threadIdx.x) >> 5);
  int lane = (int)(threadIdx.x & 31);
  int nb = wave << 6;                       // 4 consecutive 16-wide tiles
  if (nb >= N) return;
  int half = lane >> 4, l16 = lane & 15, kb = half * 8;

  v8f acc0, acc1, acc2, acc3;
#pragma unroll
  for (int r = 0; r < 8; ++r) { acc0[r] = 0.0f; acc1[r] = 0.0f; acc2[r] = 0.0f; acc3[r] = 0.0f; }

  const __bf16* arow = A + (long)l16 * a_row_stride;
  gemm_panel4(arow,
              W + (long)(nb + 0 * 16 + l16) * K, W + (long)(nb + 1 * 16 + l16) * K,
              W + (long)(nb + 2 * 16 + l16) * K, W + (long)(nb + 3 * 16 + l16) * K,
              K, kb, acc0, acc1, acc2, acc3);

  v8f* accs[4] = {&acc0, &acc1, &acc2, &acc3};
#pragma unroll
  for (int u = 0; u < 4; ++u) {
    int col = nb + u * 16 + l16;
    float bv = bias ? bias[col] : 0.0f;
#pragma unroll
    for (int r = 0; r < 8; ++r) {
      int m = r + half * 8;
      float v = (*accs[u])[r] + bv;
      if (act == 1) v = tanhf(v);
      long idx = (long)m * c_row_stride + c_off + col;
      if (Cf)  Cf[idx]  = v;
      if (Cbf) Cbf[idx] = (__bf16)v;
    }
  }
}

// ---------------- small helper kernels ----------------
__global__ void f32_to_bf16_kernel(const float* __restrict__ s, __bf16* __restrict__ d, long n) {
  long i = (long)blockIdx.x * blockDim.x + threadIdx.x;
  if (i < n) d[i] = (__bf16)s[i];
}

__global__ void zero_f32_kernel(float* __restrict__ p, long n) {
  long i = (long)blockIdx.x * blockDim.x + threadIdx.x;
  if (i < n) p[i] = 0.0f;
}

__global__ void zero_bf16_kernel(__bf16* __restrict__ p, long n) {
  long i = (long)blockIdx.x * blockDim.x + threadIdx.x;
  if (i < n) p[i] = (__bf16)0.0f;
}

// embeddings gathered straight into bf16 (GEMM A operand)
__global__ void embed_enc_kernel(const float* __restrict__ emb, const int* __restrict__ ids,
                                 __bf16* __restrict__ x, long n) {
  long i = (long)blockIdx.x * blockDim.x + threadIdx.x;
  if (i >= n) return;
  long tokpos = i / HDIM;
  int  h      = (int)(i % HDIM);
  x[i] = (__bf16)emb[(long)ids[tokpos] * HDIM + h];
}

// dst[b][dst_off + j] = src[b][j]
__global__ void copy_rows_f32_kernel(const float* __restrict__ src, int src_cols,
                                     float* __restrict__ dst, int dst_stride, int dst_off,
                                     long n) {
  long i = (long)blockIdx.x * blockDim.x + threadIdx.x;
  if (i >= n) return;
  int b = (int)(i / src_cols), j = (int)(i % src_cols);
  dst[(long)b * dst_stride + dst_off + j] = src[i];
}

__global__ void copy_rows_bf16_kernel(const __bf16* __restrict__ src, int src_cols,
                                      __bf16* __restrict__ dst, int dst_stride, int dst_off,
                                      long n) {
  long i = (long)blockIdx.x * blockDim.x + threadIdx.x;
  if (i >= n) return;
  int b = (int)(i / src_cols), j = (int)(i % src_cols);
  dst[(long)b * dst_stride + dst_off + j] = src[i];
}

// decin[b] = concat(dec_emb[prev_tok(b,t)], op[b])   (bf16, feeds layer-0 gate GEMM)
__global__ void assemble_decin_kernel(const float* __restrict__ dec_emb,
                                      const int* __restrict__ out_ids,
                                      const __bf16* __restrict__ op, int t,
                                      __bf16* __restrict__ decin) {
  int i = blockIdx.x * blockDim.x + threadIdx.x;
  if (i >= BB * 2 * HDIM) return;
  int b = i / (2 * HDIM), k = i % (2 * HDIM);
  __bf16 v;
  if (k < HDIM) {
    int tok = (t == 0) ? SOS_TOK : out_ids[b * TT + (t - 1)];
    v = (__bf16)dec_emb[(long)tok * HDIM + k];
  } else {
    v = op[(long)b * HDIM + (k - HDIM)];
  }
  decin[i] = v;
}

// cat[b] = concat(cc[b], decout[b])   (bf16, feeds linout GEMM)
__global__ void assemble_cat_kernel(const float* __restrict__ cc,
                                    const __bf16* __restrict__ decout,
                                    __bf16* __restrict__ cat) {
  int i = blockIdx.x * blockDim.x + threadIdx.x;
  if (i >= BB * 2 * HDIM) return;
  int b = i / (2 * HDIM), k = i % (2 * HDIM);
  cat[i] = (k < HDIM) ? (__bf16)cc[(long)b * HDIM + k]
                      : decout[(long)b * HDIM + (k - HDIM)];
}

// sc[b][s] = dot(enc[b][s][:], q[b][:])
__global__ void attn_scores_kernel(const float* __restrict__ enc, const float* __restrict__ q,
                                   float* __restrict__ sc) {
  int i = blockIdx.x * blockDim.x + threadIdx.x;
  if (i >= BB * SS) return;
  int b = i / SS, s = i % SS;
  const float* e  = enc + ((long)b * SS + s) * HDIM;
  const float* qq = q + (long)b * HDIM;
  float acc = 0.0f;
  for (int h = 0; h < HDIM; ++h) acc += e[h] * qq[h];
  sc[i] = acc;
}

__global__ void softmax_kernel(const float* __restrict__ sc, float* __restrict__ w) {
  __shared__ float red[SS];
  int b = blockIdx.x, s = threadIdx.x;
  float v = sc[b * SS + s];
  red[s] = v; __syncthreads();
  for (int off = SS / 2; off > 0; off >>= 1) {
    if (s < off) red[s] = fmaxf(red[s], red[s + off]);
    __syncthreads();
  }
  float mx = red[0]; __syncthreads();
  float e = __expf(v - mx);
  red[s] = e; __syncthreads();
  for (int off = SS / 2; off > 0; off >>= 1) {
    if (s < off) red[s] += red[s + off];
    __syncthreads();
  }
  w[b * SS + s] = e / red[0];
}

// cc[b][h] = sum_s w[b][s] * enc[b][s][h]
__global__ void context_kernel(const float* __restrict__ w, const float* __restrict__ enc,
                               float* __restrict__ cc) {
  int i = blockIdx.x * blockDim.x + threadIdx.x;
  if (i >= BB * HDIM) return;
  int b = i / HDIM, h = i % HDIM;
  float acc = 0.0f;
  for (int s = 0; s < SS; ++s) acc += w[b * SS + s] * enc[((long)b * SS + s) * HDIM + h];
  cc[i] = acc;
}

// ---------------- host orchestration ----------------
static inline size_t alignup256(size_t x) { return (x + 255) & ~(size_t)255; }

extern "C" void kernel_launch(void* const* d_in, const int* in_sizes, int n_in,
                              void* d_out, int out_size, void* d_ws, size_t ws_size,
                              hipStream_t stream) {
  (void)in_sizes; (void)n_in; (void)out_size; (void)ws_size;

  const float* enc_emb  = (const float*)d_in[0];
  const float* dec_emb  = (const float*)d_in[1];
  const float* enc_Wih  = (const float*)d_in[2];   // [L,2,2048,1024]
  const float* enc_Whh  = (const float*)d_in[3];   // [L,2,2048,512]
  const float* enc_bias = (const float*)d_in[4];   // [L,2,2048]
  const float* dec_Wih0 = (const float*)d_in[5];   // [4096,2048]
  const float* dec_WihR = (const float*)d_in[6];   // [1,4096,1024]
  const float* dec_Whh  = (const float*)d_in[7];   // [L,4096,1024]
  const float* dec_bias = (const float*)d_in[8];   // [L,4096]
  const float* linin_W  = (const float*)d_in[9];   // [1024,1024]
  const float* linout_W = (const float*)d_in[10];  // [1024,2048]
  const float* gen_W    = (const float*)d_in[11];  // [32000,1024]
  const float* gen_b    = (const float*)d_in[12];  // [32000]
  const int*   inp      = (const int*)d_in[13];    // [B,S]
  const int*   out_ids  = (const int*)d_in[14];    // [B,T]
  float*       out      = (float*)d_out;           // [B,T,V]

  // ---- workspace layout ----
  char* p = (char*)d_ws;
  auto alloc_bf = [&](size_t elems) { __bf16* r = (__bf16*)p; p += alignup256(elems * 2); return r; };
  auto alloc_f  = [&](size_t elems) { float*  r = (float*)p;  p += alignup256(elems * 4); return r; };

  const size_t n_encWih  = (size_t)LLAY * 2 * 4 * HIDE * HDIM;
  const size_t n_encWhh  = (size_t)LLAY * 2 * 4 * HIDE * HIDE;
  const size_t n_decWih0 = (size_t)4 * HDIM * 2 * HDIM;
  const size_t n_decWihR = (size_t)4 * HDIM * HDIM;
  const size_t n_decWhh  = (size_t)LLAY * 4 * HDIM * HDIM;
  const size_t n_linin   = (size_t)HDIM * HDIM;
  const size_t n_linout  = (size_t)HDIM * 2 * HDIM;
  const size_t n_gen     = (size_t)VV * HDIM;

  __bf16* w_encWih  = alloc_bf(n_encWih);
  __bf16* w_encWhh  = alloc_bf(n_encWhh);
  __bf16* w_decWih0 = alloc_bf(n_decWih0);
  __bf16* w_decWihR = alloc_bf(n_decWihR);
  __bf16* w_decWhh  = alloc_bf(n_decWhh);
  __bf16* w_linin   = alloc_bf(n_linin);
  __bf16* w_linout  = alloc_bf(n_linout);
  __bf16* w_gen     = alloc_bf(n_gen);

  const size_t BSH = (size_t)BB * SS * HDIM;
  __bf16* xbfA    = alloc_bf(BSH);                 // bf16 activation sequences
  __bf16* xbfB    = alloc_bf(BSH);
  float*  enc_f32 = alloc_f(BSH);                  // f32 encoder output for attention
  __bf16* eh0     = alloc_bf((size_t)BB * HIDE);   // encoder h ping-pong (bf16)
  __bf16* eh1     = alloc_bf((size_t)BB * HIDE);
  float*  ecs     = alloc_f((size_t)BB * HIDE);    // encoder c (f32)
  __bf16* dh0a    = alloc_bf((size_t)BB * HDIM);   // decoder h ping-pong (bf16)
  __bf16* dh0b    = alloc_bf((size_t)BB * HDIM);
  __bf16* dh1a    = alloc_bf((size_t)BB * HDIM);
  __bf16* dh1b    = alloc_bf((size_t)BB * HDIM);
  float*  dc0     = alloc_f((size_t)BB * HDIM);    // decoder c (f32)
  float*  dc1     = alloc_f((size_t)BB * HDIM);
  __bf16* op_bf   = alloc_bf((size_t)BB * HDIM);   // attention output feedback (bf16)
  float*  qb      = alloc_f((size_t)BB * HDIM);
  float*  ccb     = alloc_f((size_t)BB * HDIM);
  __bf16* decin   = alloc_bf((size_t)BB * 2 * HDIM);
  __bf16* catb    = alloc_bf((size_t)BB * 2 * HDIM);
  float*  scb     = alloc_f((size_t)BB * SS);
  float*  wtb     = alloc_f((size_t)BB * SS);

  auto conv = [&](const float* s, __bf16* d, size_t n) {
    f32_to_bf16_kernel<<<(unsigned)((n + 255) / 256), 256, 0, stream>>>(s, d, (long)n);
  };

  // ---- weights -> bf16 (rebuilt every call; deterministic) ----
  conv(enc_Wih,  w_encWih,  n_encWih);
  conv(enc_Whh,  w_encWhh,  n_encWhh);
  conv(dec_Wih0, w_decWih0, n_decWih0);
  conv(dec_WihR, w_decWihR, n_decWihR);
  conv(dec_Whh,  w_decWhh,  n_decWhh);
  conv(linin_W,  w_linin,   n_linin);
  conv(linout_W, w_linout,  n_linout);
  conv(gen_W,    w_gen,     n_gen);

  // ---- encoder ----
  embed_enc_kernel<<<(unsigned)((BSH + 255) / 256), 256, 0, stream>>>(enc_emb, inp, xbfA,
                                                                      (long)BSH);

  __bf16* dh_cur[2] = {dh0a, dh1a};
  float*  dcst[2]   = {dc0, dc1};
  __bf16* xcur = xbfA;
  __bf16* xnxt = xbfB;
  for (int l = 0; l < LLAY; ++l) {
    int last = (l == LLAY - 1);
    for (int dir = 0; dir < 2; ++dir) {
      zero_bf16_kernel<<<(BB * HIDE) / 256, 256, 0, stream>>>(eh0, (long)BB * HIDE);
      zero_f32_kernel<<<(BB * HIDE) / 256, 256, 0, stream>>>(ecs, (long)BB * HIDE);
      __bf16* hp = eh0; __bf16* hn = eh1;
      const __bf16* Wih = w_encWih + (size_t)(l * 2 + dir) * 4 * HIDE * HDIM;
      const __bf16* Whh = w_encWhh + (size_t)(l * 2 + dir) * 4 * HIDE * HIDE;
      const float*  bia = enc_bias + (size_t)(l * 2 + dir) * 4 * HIDE;
      for (int stp = 0; stp < SS; ++stp) {
        int t = dir ? (SS - 1 - stp) : stp;
        lstm_step_kernel<<<HIDE / 64, 128, 0, stream>>>(
            xcur + (long)t * HDIM, (long)SS * HDIM, HDIM,
            hp, HIDE, Wih, Whh, bia, ecs, hn,
            last ? nullptr : (xnxt + (long)t * HDIM + dir * HIDE),
            last ? (enc_f32 + (long)t * HDIM + dir * HIDE) : nullptr,
            (long)SS * HDIM);
        __bf16* tmp = hp; hp = hn; hn = tmp;
      }
      // final states -> decoder init (concat fwd/bwd halves)
      copy_rows_bf16_kernel<<<(BB * HIDE) / 256, 256, 0, stream>>>(
          hp, HIDE, dh_cur[l], HDIM, dir * HIDE, (long)BB * HIDE);
      copy_rows_f32_kernel<<<(BB * HIDE) / 256, 256, 0, stream>>>(
          ecs, HIDE, dcst[l], HDIM, dir * HIDE, (long)BB * HIDE);
    }
    __bf16* tmp = xcur; xcur = xnxt; xnxt = tmp;
  }
  float* enc = enc_f32;  // [B,S,H] f32 for attention VALU kernels

  // ---- decoder with attention ----
  zero_bf16_kernel<<<(BB * HDIM) / 256, 256, 0, stream>>>(op_bf, (long)BB * HDIM);
  __bf16* hA0 = dh0a; __bf16* hB0 = dh0b;
  __bf16* hA1 = dh1a; __bf16* hB1 = dh1b;

  for (int t = 0; t < TT; ++t) {
    assemble_decin_kernel<<<(BB * 2 * HDIM) / 256, 256, 0, stream>>>(dec_emb, out_ids, op_bf,
                                                                     t, decin);
    // layer 0: in = [B,2H], hid = H
    lstm_step_kernel<<<HDIM / 64, 128, 0, stream>>>(
        decin, (long)(2 * HDIM), 2 * HDIM, hA0, HDIM,
        w_decWih0, w_decWhh, dec_bias, dc0, hB0, nullptr, nullptr, 0);
    // layer 1: in = h of layer 0
    lstm_step_kernel<<<HDIM / 64, 128, 0, stream>>>(
        hB0, (long)HDIM, HDIM, hA1, HDIM,
        w_decWihR, w_decWhh + (size_t)4 * HDIM * HDIM, dec_bias + 4 * HDIM,
        dc1, hB1, nullptr, nullptr, 0);
    __bf16* decout = hB1;

    // attention: q = decout @ linin^T  (f32 out, feeds VALU dot kernels)
    gemm16x4_kernel<<<HDIM / 256, 128, 0, stream>>>(decout, (long)HDIM, HDIM, w_linin,
                                                    nullptr, qb, nullptr, (long)HDIM, 0,
                                                    HDIM, 0);
    attn_scores_kernel<<<(BB * SS + 127) / 128, 128, 0, stream>>>(enc, qb, scb);
    softmax_kernel<<<BB, SS, 0, stream>>>(scb, wtb);
    context_kernel<<<(BB * HDIM) / 128, 128, 0, stream>>>(wtb, enc, ccb);
    assemble_cat_kernel<<<(BB * 2 * HDIM) / 256, 256, 0, stream>>>(ccb, decout, catb);
    // op = tanh(cat @ linout^T)  (bf16 out, feeds decin + generator)
    gemm16x4_kernel<<<HDIM / 256, 128, 0, stream>>>(catb, (long)(2 * HDIM), 2 * HDIM,
                                                    w_linout, nullptr, nullptr, op_bf,
                                                    (long)HDIM, 0, HDIM, 1 /*tanh*/);
    // generator: logits -> d_out[:, t, :]  (f32 + bias)
    gemm16x4_kernel<<<VV / 256, 128, 0, stream>>>(op_bf, (long)HDIM, HDIM, w_gen, gen_b,
                                                  out, nullptr, (long)TT * VV, (long)t * VV,
                                                  VV, 0);
    { __bf16* tmp = hA0; hA0 = hB0; hB0 = tmp; }
    { __bf16* tmp = hA1; hA1 = hB1; hB1 = tmp; }
  }
}
